// DeltaNet_22488448762207
// MI455X (gfx1250) — matrix operand order repaired and checked by reading, not compiled, not run on hardware
//
#include <hip/hip_runtime.h>
#include <hip/hip_bf16.h>
#include <math.h>

// ---------------- problem constants ----------------
#define B_  4
#define T_  2048
#define HS_ 1024
#define H_  16
#define R_  16
#define D_  64
#define M_  (B_*T_)      // 8192 rows
#define CC  64           // scan chunk length

typedef __bf16 bf16_t;
typedef __attribute__((ext_vector_type(16))) __bf16 v16bf;
typedef __attribute__((ext_vector_type(8)))  __bf16 v8bf;
typedef __attribute__((ext_vector_type(8)))  float  v8f;
typedef __attribute__((ext_vector_type(4))) unsigned int u32x4;
typedef __attribute__((ext_vector_type(8))) int          i32x8;
typedef __attribute__((ext_vector_type(4))) int          i32x4;

// Tensor Data Mover availability.  This toolchain (clang-23 / therock-10.0)
// uses the 6-arg builtin: (u32x4 g0, i32x8 g1, i32x4, i32x4, i32x8, i32 cpol).
#if defined(__gfx1250__) && __has_builtin(__builtin_amdgcn_tensor_load_to_lds) && \
    __has_builtin(__builtin_amdgcn_s_wait_tensorcnt)
#define USE_TDM 1
#else
#define USE_TDM 0
#endif

// ---------------- WMMA helpers (CDNA5 wave32) ----------------
__device__ __forceinline__ v8f wmma_bf16_f32(v16bf a, v16bf b, v8f c) {
  // D = A(16x32 bf16) * B(32x16 bf16) + C(16x16 f32)
  return __builtin_amdgcn_wmma_f32_16x16x32_bf16(false, a, false, b, (short)0, c,
                                                 false, false);
}

// A fragment (16x32 bf16): lane m (0-15) holds row m, K = {0..7, 16..23};
// lane m+16 holds row m, K = {8..15, 24..31}.  `row` points at the row start.
__device__ __forceinline__ v16bf load_frag_a(const bf16_t* row, int lane) {
  int kb = (lane >> 4) * 8;
  v16bf f;
#pragma unroll
  for (int i = 0; i < 8; ++i) { f[i] = row[kb + i]; f[i + 8] = row[kb + 16 + i]; }
  return f;
}

// B fragment (32x16 bf16), column-contiguous storage [N][K]:
// lane n (0-15) holds column n, K = {0..7, 16..23}; lane n+16 holds K = {8..15, 24..31}.
__device__ __forceinline__ v16bf load_frag_b(const bf16_t* col, int lane) {
  int kb = (lane >> 4) * 8;
  v16bf f;
#pragma unroll
  for (int i = 0; i < 8; ++i) { f[i] = col[kb + i]; f[i + 8] = col[kb + 16 + i]; }
  return f;
}

__device__ __forceinline__ v8f zero8() {
  v8f z;
#pragma unroll
  for (int i = 0; i < 8; ++i) z[i] = 0.0f;
  return z;
}

#if USE_TDM
// Issue one TDM 2D-tile load: tile_rows x tile_cols bf16 elements, row pitch
// `stride_elems`, into LDS at byte offset `lds_off` (row-major, contiguous).
// D# layout per CDNA5 ISA (group0 128b, group1 256b).
__device__ __forceinline__ void tdm_load_2d(const bf16_t* gbase, unsigned lds_off,
                                            unsigned tile_rows, unsigned tile_cols,
                                            unsigned stride_elems)
{
  unsigned long long ga = (unsigned long long)(const void*)gbase;
  u32x4 g0;
  g0.x = 1u;                                                 // count=1, user D#
  g0.y = lds_off;                                            // lds_addr (bytes)
  g0.z = (unsigned)(ga & 0xFFFFFFFFu);                       // global_addr[31:0]
  g0.w = (unsigned)((ga >> 32) & 0x01FFFFFFu) | (2u << 30);  // addr[56:32], type=2
  i32x8 g1;
  g1[0] = (int)(1u << 16);                                   // data_size=1 -> 2B
  g1[1] = (int)((tile_cols & 0xFFFFu) << 16);                // tensor_dim0[15:0]
  g1[2] = (int)((tile_cols >> 16) | ((tile_rows & 0xFFFFu) << 16)); // dim0 hi | dim1 lo
  g1[3] = (int)((tile_rows >> 16) | ((tile_cols & 0xFFFFu) << 16)); // dim1 hi | tile_dim0
  g1[4] = (int)(tile_rows & 0xFFFFu);                        // tile_dim1 (tile_dim2=0)
  g1[5] = (int)stride_elems;                                 // tensor_dim0_stride[31:0]
  g1[6] = 0;                                                 // stride0 hi | dim1_stride lo
  g1[7] = 0;
  i32x4 z4 = {0, 0, 0, 0};
  i32x8 z8 = {0, 0, 0, 0, 0, 0, 0, 0};
  // 6-arg form (clang-23 / therock-10.0 lane)
  __builtin_amdgcn_tensor_load_to_lds(g0, g1, z4, z4, z8, 0);
}
#endif

// ---------------- conversion / small kernels ----------------
__global__ void cvt_bf16_kernel(const float* __restrict__ in, bf16_t* __restrict__ out, int n) {
  for (int i = blockIdx.x * blockDim.x + threadIdx.x; i < n; i += gridDim.x * blockDim.x)
    out[i] = (bf16_t)in[i];
}

__global__ void rope_tables_kernel(float* __restrict__ cosT, float* __restrict__ sinT) {
  int t = blockIdx.x;          // 0..T-1
  int j = threadIdx.x;         // 0..31  (half = D/2 = 32)
  float inv = powf(10000.0f, -(float)j / 32.0f);
  float f = (float)t * inv;
  cosT[t * 32 + j] = cosf(f);
  sinT[t * 32 + j] = sinf(f);
}

__global__ void sigmoid_kernel(const float* __restrict__ in, float* __restrict__ out, int n) {
  int i = blockIdx.x * blockDim.x + threadIdx.x;
  if (i < n) out[i] = 1.0f / (1.0f + expf(-in[i]));
}

// ---------------- tiled WMMA GEMM: C[M,N] = A[M,K] * W[N,K]^T (+bias +resid) ----
// workgroup = 256 threads = 8 waves; tile 64(M) x 128(N); K step 32.
// TDM path: double-buffered LDS tiles filled by the Tensor Data Mover (wave 0
// issues descriptors; DMA overlaps with WMMA on the other buffer).
__global__ __launch_bounds__(256) void gemm_bf16_kernel(
    const bf16_t* __restrict__ A, const bf16_t* __restrict__ W,
    float* __restrict__ C, int N, int K,
    const float* __restrict__ bias, const float* __restrict__ resid)
{
  __shared__ bf16_t As[2][64][32];
  __shared__ bf16_t Bs[2][128][32];

  const int tid  = threadIdx.x;
  const int wid  = tid >> 5;
  const int lane = tid & 31;
  const int m0 = blockIdx.x * 64;
  const int n0 = blockIdx.y * 128;
  const int wm = (wid >> 1) * 16;   // wave's M offset in tile (0..48)
  const int wn = (wid & 1) * 64;    // wave's N offset in tile (0 or 64)

  v8f acc[4];
#pragma unroll
  for (int i = 0; i < 4; ++i) acc[i] = zero8();

  int cur = 0;

#if USE_TDM
  const unsigned ldsA0 = (unsigned)(size_t)&As[0][0][0];
  const unsigned ldsA1 = (unsigned)(size_t)&As[1][0][0];
  const unsigned ldsB0 = (unsigned)(size_t)&Bs[0][0][0];
  const unsigned ldsB1 = (unsigned)(size_t)&Bs[1][0][0];
  const bf16_t* Abase = A + (size_t)m0 * K;
  const bf16_t* Wbase = W + (size_t)n0 * K;

  if (wid == 0) {   // prologue DMA into buffer 0
    tdm_load_2d(Abase, ldsA0, 64, 32, (unsigned)K);
    tdm_load_2d(Wbase, ldsB0, 128, 32, (unsigned)K);
    __builtin_amdgcn_s_wait_tensorcnt(0);
  }
  __syncthreads();

  for (int k0 = 0; k0 < K; k0 += 32) {
    const bool more = (k0 + 32) < K;
    if (wid == 0 && more) {         // DMA next tiles while we compute
      tdm_load_2d(Abase + k0 + 32, cur ? ldsA0 : ldsA1, 64, 32, (unsigned)K);
      tdm_load_2d(Wbase + k0 + 32, cur ? ldsB0 : ldsB1, 128, 32, (unsigned)K);
    }
    v16bf afrag = load_frag_a(&As[cur][wm + (lane & 15)][0], lane);
#pragma unroll
    for (int nt = 0; nt < 4; ++nt) {
      v16bf bfrag = load_frag_b(&Bs[cur][wn + nt * 16 + (lane & 15)][0], lane);
      acc[nt] = wmma_bf16_f32(afrag, bfrag, acc[nt]);
    }
    if (wid == 0 && more) __builtin_amdgcn_s_wait_tensorcnt(0);
    __syncthreads();
    cur ^= 1;
  }
#else
  // fallback: cooperative VGPR staging (no TDM builtin on this toolchain)
  const int ar = tid >> 2, ac = (tid & 3) * 8;     // A tile: 64 rows x 4 chunks
  const int br = tid >> 1, bc = (tid & 1) * 16;    // B tile: 128 rows x 2 chunks
  for (int k0 = 0; k0 < K; k0 += 32) {
    const bf16_t* sa = A + (size_t)(m0 + ar) * K + k0 + ac;
    *(v8bf*)&As[0][ar][ac] = *(const v8bf*)sa;
    const bf16_t* sb = W + (size_t)(n0 + br) * K + k0 + bc;
    *(v8bf*)&Bs[0][br][bc]     = *(const v8bf*)sb;
    *(v8bf*)&Bs[0][br][bc + 8] = *(const v8bf*)(sb + 8);
    if (k0 + 64 < K) {
      __builtin_prefetch(A + (size_t)(m0 + ar) * K + k0 + 64 + ac, 0, 1);
      __builtin_prefetch(W + (size_t)(n0 + br) * K + k0 + 64 + bc, 0, 1);
    }
    __syncthreads();
    v16bf afrag = load_frag_a(&As[0][wm + (lane & 15)][0], lane);
#pragma unroll
    for (int nt = 0; nt < 4; ++nt) {
      v16bf bfrag = load_frag_b(&Bs[0][wn + nt * 16 + (lane & 15)][0], lane);
      acc[nt] = wmma_bf16_f32(afrag, bfrag, acc[nt]);
    }
    __syncthreads();
  }
#endif

  // epilogue: C layout — vgpr i holds row (i + 8*(lane>>4)), col (lane&15)
  const int rbase = m0 + wm + ((lane >> 4) << 3);
  const int cbase = n0 + wn + (lane & 15);
#pragma unroll
  for (int nt = 0; nt < 4; ++nt) {
    int ccol = cbase + nt * 16;
#pragma unroll
    for (int i = 0; i < 8; ++i) {
      int row = rbase + i;
      float v = acc[nt][i];
      if (bias)  v += bias[ccol];
      if (resid) v += resid[(size_t)row * N + ccol];
      C[(size_t)row * N + ccol] = v;
    }
  }
}

// ---------------- RoPE + rank projection + gate --------------------------------
// one wave per (b,t,h); lane l owns dim pair (2l, 2l+1)
__global__ __launch_bounds__(32) void rope_qr_kernel(
    const float* __restrict__ qg, const float* __restrict__ kg,
    const float* __restrict__ glog,
    const float* __restrict__ cosT, const float* __restrict__ sinT,
    const float* __restrict__ Wq, const float* __restrict__ Wk,
    float* __restrict__ q_r, float* __restrict__ a_out)
{
  int bid = blockIdx.x;
  int h = bid & (H_ - 1);
  int t = (bid >> 4) & (T_ - 1);
  int b = bid >> 15;
  int l = threadIdx.x;

  size_t base = (size_t)(b * T_ + t) * HS_ + h * D_;
  float c = cosT[t * 32 + l], s = sinT[t * 32 + l];
  float q1 = qg[base + 2 * l], q2 = qg[base + 2 * l + 1];
  float k1 = kg[base + 2 * l], k2 = kg[base + 2 * l + 1];
  float qe = q1 * c - q2 * s, qo = q1 * s + q2 * c;
  float ke = k1 * c - k2 * s, ko = k1 * s + k2 * c;

  for (int r = 0; r < R_; ++r) {
    const float* wq = Wq + ((size_t)h * R_ + r) * D_;
    const float* wk = Wk + ((size_t)h * R_ + r) * D_;
    float pq = qe * wq[2 * l] + qo * wq[2 * l + 1];
    float pk = ke * wk[2 * l] + ko * wk[2 * l + 1];
#pragma unroll
    for (int off = 16; off > 0; off >>= 1) {
      pq += __shfl_down(pq, off, 32);
      pk += __shfl_down(pk, off, 32);
    }
    if (l == 0) {
      size_t o = ((((size_t)b * H_ + h) * T_) + t) * R_ + r;
      q_r[o] = pq;
      float gl = glog[(size_t)(b * T_ + t) * (H_ * R_) + h * R_ + r]; // g_b fused in GEMM
      float g = 1.0f / (1.0f + expf(-gl));
      a_out[o] = g * pk * 0.125f;  // scale = 1/sqrt(64)
    }
  }
}

// ---------------- chunked WMMA scan --------------------------------------------
// one wave per (b,h). State M (16r x 64d) fp32 kept in 4 C-fragments.
// Chunk math:  W[i,s]=Σ_r (q λ^i)(a λ^{-s}),  y = (W⊙causal)V + (q λ^{i+1})M_prev,
//              M ← λ^C M + (a λ^{C-1-s})ᵀ V
__global__ __launch_bounds__(32) void scan_kernel(
    const float* __restrict__ q_r, const float* __restrict__ a_in,
    const float* __restrict__ vg, const float* __restrict__ decay,
    bf16_t* __restrict__ Y)
{
  const int b = blockIdx.x >> 4, h = blockIdx.x & 15;
  const int lane = threadIdx.x;
  const int lrow = lane & 15, lhi = (lane >> 4) << 3;

  __shared__ float  pw [CC + 1][16];   // decay^i
  __shared__ float  ipw[CC][16];       // decay^{-i}
  __shared__ bf16_t qt [CC][32];       // q*λ^i      (A op, K=16 padded to 32)
  __shared__ bf16_t qd [CC][32];       // q*λ^{i+1}  (A op, padded)
  __shared__ bf16_t atB[CC][32];       // a*λ^{-s}   (B op layout [n=s][k=r], padded)
  __shared__ bf16_t ahat[16][64];      // a*λ^{C-1-s}(A op layout [r][s], K=64)
  __shared__ bf16_t vT [64][64];       // V^T        (B op layout [n=d][k=s], K=64)
  __shared__ bf16_t mpB[64][32];       // M_prev     (B op layout [n=d][k=r], padded)
  __shared__ bf16_t sA [16][64];       // masked S   (A op layout [i][s], K=64)

  if (lane < 16) {
    float dc = decay[h * 16 + lane];
    float p = 1.0f;
    for (int i = 0; i <= CC; ++i) {
      pw[i][lane] = p;
      if (i < CC) ipw[i][lane] = 1.0f / p;
      p *= dc;
    }
  }
  const bf16_t zb = (bf16_t)0.0f;
  for (int i = lane; i < CC; i += 32)
    for (int r = 16; r < 32; ++r) { qt[i][r] = zb; qd[i][r] = zb; atB[i][r] = zb; }
  for (int d = lane; d < 64; d += 32)
    for (int r = 16; r < 32; ++r) mpB[d][r] = zb;
  __syncthreads();

  v8f Ms[4];
#pragma unroll
  for (int nt = 0; nt < 4; ++nt) Ms[nt] = zero8();

  const size_t qa_base = (((size_t)b * H_ + h) * T_) * R_;

  for (int ch = 0; ch < T_ / CC; ++ch) {
    const int t0 = ch * CC;
    // ---- stage chunk operands (bf16, fragment-friendly layouts) ----
    for (int i = lane; i < CC; i += 32) {
      const float* qp = q_r + qa_base + (size_t)(t0 + i) * R_;
      const float* ap = a_in + qa_base + (size_t)(t0 + i) * R_;
#pragma unroll
      for (int r = 0; r < 16; ++r) {
        float q = qp[r], a = ap[r];
        qt[i][r]   = (bf16_t)(q * pw[i][r]);
        qd[i][r]   = (bf16_t)(q * pw[i + 1][r]);
        atB[i][r]  = (bf16_t)(a * ipw[i][r]);
        ahat[r][i] = (bf16_t)(a * pw[CC - 1 - i][r]);
      }
    }
    for (int idx = lane; idx < CC * 64; idx += 32) {
      int i = idx >> 6, d = idx & 63;
      vT[d][i] = (bf16_t)vg[(size_t)(b * T_ + t0 + i) * HS_ + h * D_ + d];
    }
    // spill M_prev fragments to LDS in B-operand layout
#pragma unroll
    for (int nt = 0; nt < 4; ++nt) {
      int d = nt * 16 + lrow;
#pragma unroll
      for (int i = 0; i < 8; ++i) mpB[d][i + lhi] = (bf16_t)Ms[nt][i];
    }
    __syncthreads();

    v16bf vfr[4][2], mfr[4], ahfr[2];
#pragma unroll
    for (int nt = 0; nt < 4; ++nt) {
      vfr[nt][0] = load_frag_b(&vT[nt * 16 + lrow][0],  lane);
      vfr[nt][1] = load_frag_b(&vT[nt * 16 + lrow][32], lane);
      mfr[nt]    = load_frag_b(&mpB[nt * 16 + lrow][0], lane);
    }
    ahfr[0] = load_frag_a(&ahat[lrow][0],  lane);
    ahfr[1] = load_frag_a(&ahat[lrow][32], lane);

    // ---- state update: M_new = λ^C ∘ M + ahat × V ----
    v8f Mn[4];
#pragma unroll
    for (int nt = 0; nt < 4; ++nt) {
      v8f m;
#pragma unroll
      for (int i = 0; i < 8; ++i) m[i] = Ms[nt][i] * pw[CC][i + lhi];
      m = wmma_bf16_f32(ahfr[0], vfr[nt][0], m);
      m = wmma_bf16_f32(ahfr[1], vfr[nt][1], m);
      Mn[nt] = m;
    }

    // ---- outputs, one 16-row M-tile at a time ----
    for (int mt = 0; mt < 4; ++mt) {
      v16bf qtfr = load_frag_a(&qt[mt * 16 + lrow][0], lane);
      v16bf qdfr = load_frag_a(&qd[mt * 16 + lrow][0], lane);
      // S = q̃ ãᵀ, causal-masked, restaged as A operand
#pragma unroll
      for (int st = 0; st < 4; ++st) {
        v16bf atfr = load_frag_b(&atB[st * 16 + lrow][0], lane);
        v8f S = zero8();
        S = wmma_bf16_f32(qtfr, atfr, S);
#pragma unroll
        for (int i = 0; i < 8; ++i) {
          int gi = mt * 16 + i + lhi;
          int gs = st * 16 + lrow;
          sA[i + lhi][st * 16 + lrow] = (bf16_t)((gs <= gi) ? S[i] : 0.0f);
        }
      }
      __syncthreads();
      v16bf sfr0 = load_frag_a(&sA[lrow][0],  lane);
      v16bf sfr1 = load_frag_a(&sA[lrow][32], lane);
#pragma unroll
      for (int nt = 0; nt < 4; ++nt) {
        v8f y = zero8();
        y = wmma_bf16_f32(qdfr, mfr[nt], y);          // inter-chunk
        y = wmma_bf16_f32(sfr0, vfr[nt][0], y);       // intra-chunk
        y = wmma_bf16_f32(sfr1, vfr[nt][1], y);
#pragma unroll
        for (int i = 0; i < 8; ++i) {
          int gi = t0 + mt * 16 + i + lhi;
          int gd = nt * 16 + lrow;
          Y[(size_t)(b * T_ + gi) * HS_ + h * D_ + gd] = (bf16_t)y[i];
        }
      }
      __syncthreads();
    }
#pragma unroll
    for (int nt = 0; nt < 4; ++nt) Ms[nt] = Mn[nt];
    __syncthreads();
  }
}

// ---------------- LayerNorm ----------------------------------------------------
__global__ __launch_bounds__(256) void ln_kernel(
    const float* __restrict__ hres, const float* __restrict__ ln_g,
    const float* __restrict__ ln_b, float* __restrict__ out)
{
  const int row = blockIdx.x;
  const float* x = hres + (size_t)row * HS_;
  float s = 0.0f, s2 = 0.0f;
  for (int i = threadIdx.x; i < HS_; i += 256) { float v = x[i]; s += v; s2 += v * v; }
  __shared__ float rs[256], rs2[256];
  rs[threadIdx.x] = s; rs2[threadIdx.x] = s2;
  __syncthreads();
  for (int off = 128; off > 0; off >>= 1) {
    if (threadIdx.x < off) { rs[threadIdx.x] += rs[threadIdx.x + off];
                             rs2[threadIdx.x] += rs2[threadIdx.x + off]; }
    __syncthreads();
  }
  float mu  = rs[0] * (1.0f / HS_);
  float var = rs2[0] * (1.0f / HS_) - mu * mu;
  float inv = rsqrtf(var + 1e-5f);
  for (int i = threadIdx.x; i < HS_; i += 256)
    out[(size_t)row * HS_ + i] = (x[i] - mu) * inv * ln_g[i] + ln_b[i];
}

// ---------------- launch ----------------------------------------------------
extern "C" void kernel_launch(void* const* d_in, const int* in_sizes, int n_in,
                              void* d_out, int out_size, void* d_ws, size_t ws_size,
                              hipStream_t stream) {
  const float* x    = (const float*)d_in[0];
  const float* q_w  = (const float*)d_in[1];
  const float* k_w  = (const float*)d_in[2];
  const float* v_w  = (const float*)d_in[3];
  const float* Wq   = (const float*)d_in[4];
  const float* Wk   = (const float*)d_in[5];
  const float* g_w  = (const float*)d_in[6];
  const float* g_b  = (const float*)d_in[7];
  const float* bdl  = (const float*)d_in[8];
  const float* out_w = (const float*)d_in[9];
  const float* out_b = (const float*)d_in[10];
  const float* ln_g  = (const float*)d_in[11];
  const float* ln_b  = (const float*)d_in[12];
  float* out = (float*)d_out;

  char* p = (char*)d_ws;
  auto alloc = [&](size_t bytes) -> char* {
    char* r = p; p += (bytes + 255) & ~(size_t)255; return r;
  };
  bf16_t* x_bf  = (bf16_t*)alloc((size_t)M_ * HS_ * 2);
  bf16_t* qw_bf = (bf16_t*)alloc((size_t)HS_ * HS_ * 2);
  bf16_t* kw_bf = (bf16_t*)alloc((size_t)HS_ * HS_ * 2);
  bf16_t* vw_bf = (bf16_t*)alloc((size_t)HS_ * HS_ * 2);
  bf16_t* ow_bf = (bf16_t*)alloc((size_t)HS_ * HS_ * 2);
  bf16_t* gw_bf = (bf16_t*)alloc((size_t)H_ * R_ * HS_ * 2);
  float* cosT  = (float*)alloc((size_t)T_ * 32 * 4);
  float* sinT  = (float*)alloc((size_t)T_ * 32 * 4);
  float* decay = (float*)alloc((size_t)H_ * R_ * 4);
  float* q_f   = (float*)alloc((size_t)M_ * HS_ * 4);   // later aliased by Y (bf16)
  float* k_f   = (float*)alloc((size_t)M_ * HS_ * 4);   // later aliased by hres
  float* v_f   = (float*)alloc((size_t)M_ * HS_ * 4);
  float* glog  = (float*)alloc((size_t)M_ * H_ * R_ * 4);
  float* q_r   = (float*)alloc((size_t)B_ * H_ * T_ * R_ * 4);
  float* a_s   = (float*)alloc((size_t)B_ * H_ * T_ * R_ * 4);
  bf16_t* Ybf  = (bf16_t*)q_f;   // q_f dead after rope_qr
  float*  hres = k_f;            // k_f dead after rope_qr

  // 1. precision conversion
  cvt_bf16_kernel<<<2048, 256, 0, stream>>>(x,     x_bf,  M_ * HS_);
  cvt_bf16_kernel<<<1024, 256, 0, stream>>>(q_w,   qw_bf, HS_ * HS_);
  cvt_bf16_kernel<<<1024, 256, 0, stream>>>(k_w,   kw_bf, HS_ * HS_);
  cvt_bf16_kernel<<<1024, 256, 0, stream>>>(v_w,   vw_bf, HS_ * HS_);
  cvt_bf16_kernel<<<1024, 256, 0, stream>>>(out_w, ow_bf, HS_ * HS_);
  cvt_bf16_kernel<<<256,  256, 0, stream>>>(g_w,   gw_bf, H_ * R_ * HS_);

  // 2. tables
  rope_tables_kernel<<<T_, 32, 0, stream>>>(cosT, sinT);
  sigmoid_kernel<<<1, 256, 0, stream>>>(bdl, decay, H_ * R_);

  // 3. projection GEMMs (WMMA + TDM double buffering)
  dim3 g1(M_ / 64, HS_ / 128);
  gemm_bf16_kernel<<<g1, 256, 0, stream>>>(x_bf, qw_bf, q_f, HS_, HS_, nullptr, nullptr);
  gemm_bf16_kernel<<<g1, 256, 0, stream>>>(x_bf, kw_bf, k_f, HS_, HS_, nullptr, nullptr);
  gemm_bf16_kernel<<<g1, 256, 0, stream>>>(x_bf, vw_bf, v_f, HS_, HS_, nullptr, nullptr);
  dim3 g2(M_ / 64, (H_ * R_) / 128);
  gemm_bf16_kernel<<<g2, 256, 0, stream>>>(x_bf, gw_bf, glog, H_ * R_, HS_, g_b, nullptr);

  // 4. RoPE + rank projection + gate
  rope_qr_kernel<<<B_ * T_ * H_, 32, 0, stream>>>(q_f, k_f, glog, cosT, sinT, Wq, Wk, q_r, a_s);

  // 5. chunked WMMA scan
  scan_kernel<<<B_ * H_, 32, 0, stream>>>(q_r, a_s, v_f, decay, Ybf);

  // 6. output projection + bias + residual (WMMA)
  gemm_bf16_kernel<<<g1, 256, 0, stream>>>(Ybf, ow_bf, hres, HS_, HS_, out_b, x);

  // 7. LayerNorm
  ln_kernel<<<M_, 256, 0, stream>>>(hres, ln_g, ln_b, out);
}
